// MultiHeadAttention_22789096472711
// MI455X (gfx1250) — matrix operand order repaired and checked by reading, not compiled, not run on hardware
//
#include <hip/hip_runtime.h>

// ---------------------------------------------------------------------------
// Types for gfx1250 WMMA (wave32). v16bf = A/B fragment, v8f = C/D fragment.
// ---------------------------------------------------------------------------
typedef __attribute__((ext_vector_type(16))) __bf16       v16bf;
typedef __attribute__((ext_vector_type(8)))  float        v8f;
typedef __attribute__((ext_vector_type(4)))  unsigned int u32x4;

static constexpr int D_MODEL = 1024;
static constexpr int N_HEADS = 16;
static constexpr int HEADDIM = 64;
static constexpr int Bq      = 4;
static constexpr int SEQ     = 2048;
static constexpr int MROWS   = Bq * SEQ;   // 8192

__device__ __forceinline__ unsigned short f32_to_bf16(float f) {
    unsigned int u = __float_as_uint(f);
    unsigned int r = u + 0x7fffu + ((u >> 16) & 1u);   // round-to-nearest-even
    return (unsigned short)(r >> 16);
}

__device__ __forceinline__ v8f vzero8() {
    v8f v;
    #pragma unroll
    for (int i = 0; i < 8; ++i) v[i] = 0.0f;
    return v;
}

__device__ __forceinline__ v8f wmma_bf16(v16bf a, v16bf b, v8f c) {
    return __builtin_amdgcn_wmma_f32_16x16x32_bf16(
        false, a, false, b, (short)0, c, false, false);
}

// A fragment: lane group g (lane>>4) holds K = {8g..8g+7} then {16+8g..+7}.
__device__ __forceinline__ v16bf load_a_frag(const unsigned short* row, int g) {
    union { v16bf v; u32x4 u[2]; } f;
    f.u[0] = *(const u32x4*)(row + g * 8);
    f.u[1] = *(const u32x4*)(row + 16 + g * 8);
    return f.v;
}

// B fragment: lane group g holds K = {8g..8g+7} then {16+8g..+7} for column n.
__device__ __forceinline__ v16bf load_b_frag(const unsigned short* row, int g) {
    union { v16bf v; u32x4 u[2]; } f;
    f.u[0] = *(const u32x4*)(row + g * 8);
    f.u[1] = *(const u32x4*)(row + 16 + g * 8);
    return f.v;
}

// Async 16-byte global -> LDS copy (ASYNCcnt tracked, bypasses VGPRs).
__device__ __forceinline__ void async_copy_b128(const void* smem, const void* gmem) {
    unsigned int       lds = (unsigned int)(unsigned long long)(size_t)smem;
    unsigned long long ga  = (unsigned long long)(size_t)gmem;
    asm volatile("global_load_async_to_lds_b128 %0, %1, off"
                 :: "v"(lds), "v"(ga) : "memory");
}
__device__ __forceinline__ void wait_async0() {
    asm volatile("s_wait_asynccnt 0x0" ::: "memory");
}

// ---------------------------------------------------------------------------
// f32 -> bf16 conversion
// ---------------------------------------------------------------------------
__global__ void cvt_f32_bf16(const float* __restrict__ src,
                             unsigned short* __restrict__ dst, int n) {
    int i = blockIdx.x * blockDim.x + threadIdx.x;
    if (i < n) dst[i] = f32_to_bf16(src[i]);
}

// ---------------------------------------------------------------------------
// WMMA GEMM: C[M][N] = A[M][K] * Bw[N][K]^T + bias[N]
//   out_kind 0: bf16 scatter into [B*H][S][64]  (QKV projections)
//   out_kind 1: f32 row-major [M][N]            (output projection)
// Block: 128 threads (4 waves). Tile: 128x64, K-chunks of 32, double-buffered
// LDS filled with async global->LDS copies. Each wave: 32x64 (8 WMMA/chunk).
// ---------------------------------------------------------------------------
__global__ __launch_bounds__(128) void gemm_wmma(
    const unsigned short* __restrict__ A,
    const unsigned short* __restrict__ Bw,
    const float* __restrict__ bias,
    void* __restrict__ outp,
    int M, int N, int K, int Sdim, int Hh, int out_kind)
{
    __shared__ unsigned short As[2][128][40];   // 32 K + 8 pad (16B-aligned rows)
    __shared__ unsigned short Bs[2][64][40];

    const int tid  = threadIdx.x;
    const int wave = tid >> 5;
    const int lane = tid & 31;
    const int g    = lane >> 4;
    const int ln   = lane & 15;

    const int m0 = blockIdx.x * 128;
    const int n0 = blockIdx.y * 64;
    const int mw = wave * 32;

    v8f acc[2][4];
    #pragma unroll
    for (int h = 0; h < 2; ++h)
        #pragma unroll
        for (int s = 0; s < 4; ++s) acc[h][s] = vzero8();

    auto load_tiles = [&](int kc, int buf) {
        for (int i = tid; i < 512; i += 128) {          // A: 128 rows x 4 chunks
            int r = i >> 2, c = i & 3;
            async_copy_b128(&As[buf][r][c * 8],
                            &A[(size_t)(m0 + r) * K + kc + c * 8]);
        }
        for (int i = tid; i < 256; i += 128) {          // B: 64 rows x 4 chunks
            int r = i >> 2, c = i & 3;
            async_copy_b128(&Bs[buf][r][c * 8],
                            &Bw[(size_t)(n0 + r) * K + kc + c * 8]);
        }
    };

    const int nk = K >> 5;
    load_tiles(0, 0);
    wait_async0();
    __syncthreads();

    for (int kc = 0; kc < nk; ++kc) {
        const int p = kc & 1;
        if (kc + 1 < nk) load_tiles((kc + 1) << 5, p ^ 1);

        v16bf af0 = load_a_frag(&As[p][mw + ln][0],      g);
        v16bf af1 = load_a_frag(&As[p][mw + 16 + ln][0], g);
        #pragma unroll
        for (int s = 0; s < 4; ++s) {
            v16bf bf = load_b_frag(&Bs[p][s * 16 + ln][0], g);
            acc[0][s] = wmma_bf16(af0, bf, acc[0][s]);
            acc[1][s] = wmma_bf16(af1, bf, acc[1][s]);
        }
        wait_async0();
        __syncthreads();
    }

    #pragma unroll
    for (int h = 0; h < 2; ++h) {
        #pragma unroll
        for (int s = 0; s < 4; ++s) {
            #pragma unroll
            for (int r = 0; r < 8; ++r) {
                int mm = m0 + mw + h * 16 + ((lane < 16) ? r : r + 8);
                int nn = n0 + s * 16 + ln;
                float v = acc[h][s][r] + bias[nn];
                if (out_kind == 0) {
                    int b = mm / Sdim, sq = mm % Sdim;
                    int hh = nn >> 6, e = nn & 63;
                    ((unsigned short*)outp)
                        [(((size_t)(b * Hh + hh) * Sdim + sq) << 6) + e] = f32_to_bf16(v);
                } else {
                    ((float*)outp)[(size_t)mm * N + nn] = v;
                }
            }
        }
    }
}

// ---------------------------------------------------------------------------
// Flash attention: grid = (S/64, B*H), block = 128 (4 waves).
// ---------------------------------------------------------------------------
__global__ __launch_bounds__(128) void attn_wmma(
    const unsigned short* __restrict__ Qb,
    const unsigned short* __restrict__ Kb,
    const unsigned short* __restrict__ Vb,
    unsigned short* __restrict__ ctx,
    int Sdim, int Hh)
{
    __shared__ unsigned short Ks[64][72];  // [t][d]
    __shared__ unsigned short Vt[64][72];  // [e][t] (transposed)
    __shared__ unsigned short Ps[64][72];  // bf16 probabilities [m][t]
    __shared__ float Sb[64][65];
    __shared__ float mrow[64], lrow[64], arow[64];

    const int tid  = threadIdx.x;
    const int wave = tid >> 5;
    const int lane = tid & 31;
    const int g    = lane >> 4;
    const int ln   = lane & 15;

    const int qt = blockIdx.x;
    const int bh = blockIdx.y;
    const size_t base = (size_t)bh * Sdim * HEADDIM;

    const int qrow = qt * 64 + wave * 16 + ln;
    v16bf qf0 = load_a_frag(Qb + base + (size_t)qrow * HEADDIM + 0,  g);
    v16bf qf1 = load_a_frag(Qb + base + (size_t)qrow * HEADDIM + 32, g);

    v8f o[4];
    #pragma unroll
    for (int s = 0; s < 4; ++s) o[s] = vzero8();

    if (tid < 64) { mrow[tid] = -3.0e38f; lrow[tid] = 0.0f; }

    const float scale = 0.03125f;  // 1/sqrt(1024)

    for (int kt = 0; kt < Sdim / 64; ++kt) {
        __syncthreads();
        // K tile via async global->LDS
        for (int i = tid; i < 512; i += 128) {
            int r = i >> 3, c = i & 7;
            async_copy_b128(&Ks[r][c * 8],
                            &Kb[base + (size_t)(kt * 64 + r) * HEADDIM + c * 8]);
        }
        // V tile transposed into [e][t] (manual: async cannot transpose)
        for (int i = tid; i < 2048; i += 128) {
            int r = i >> 5, c = i & 31;
            unsigned int w = *(const unsigned int*)
                &Vb[base + (size_t)(kt * 64 + r) * HEADDIM + c * 2];
            Vt[c * 2 + 0][r] = (unsigned short)(w & 0xffffu);
            Vt[c * 2 + 1][r] = (unsigned short)(w >> 16);
        }
        wait_async0();
        __syncthreads();

        #pragma unroll
        for (int s = 0; s < 4; ++s) {
            v8f c = vzero8();
            c = wmma_bf16(qf0, load_b_frag(&Ks[s * 16 + ln][0],  g), c);
            c = wmma_bf16(qf1, load_b_frag(&Ks[s * 16 + ln][32], g), c);
            #pragma unroll
            for (int r = 0; r < 8; ++r) {
                int mm = wave * 16 + ((lane < 16) ? r : r + 8);
                Sb[mm][s * 16 + ln] = c[r] * scale;
            }
        }
        __syncthreads();

        if (tid < 64) {
            float mprev = mrow[tid], tmax = mprev;
            for (int j = 0; j < 64; ++j) tmax = fmaxf(tmax, Sb[tid][j]);
            float a  = __expf(mprev - tmax);
            float ps = 0.0f;
            for (int j = 0; j < 64; ++j) {
                float p = __expf(Sb[tid][j] - tmax);
                ps += p;
                Ps[tid][j] = f32_to_bf16(p);
            }
            mrow[tid] = tmax;
            lrow[tid] = lrow[tid] * a + ps;
            arow[tid] = a;
        }
        __syncthreads();

        v16bf pf0 = load_a_frag(&Ps[wave * 16 + ln][0],  g);
        v16bf pf1 = load_a_frag(&Ps[wave * 16 + ln][32], g);
        #pragma unroll
        for (int s = 0; s < 4; ++s) {
            #pragma unroll
            for (int r = 0; r < 8; ++r) {
                float f = arow[wave * 16 + ((lane < 16) ? r : r + 8)];
                o[s][r] *= f;
            }
            o[s] = wmma_bf16(pf0, load_b_frag(&Vt[s * 16 + ln][0],  g), o[s]);
            o[s] = wmma_bf16(pf1, load_b_frag(&Vt[s * 16 + ln][32], g), o[s]);
        }
    }
    __syncthreads();

    const int b = bh / Hh, h = bh % Hh;
    #pragma unroll
    for (int s = 0; s < 4; ++s) {
        #pragma unroll
        for (int r = 0; r < 8; ++r) {
            int   mloc = wave * 16 + ((lane < 16) ? r : r + 8);
            int   qg   = qt * 64 + mloc;
            float v    = o[s][r] / lrow[mloc];
            int   e    = s * 16 + ln;
            ctx[(size_t)(b * Sdim + qg) * D_MODEL + h * HEADDIM + e] = f32_to_bf16(v);
        }
    }
}

// ---------------------------------------------------------------------------
// Host launcher
// ---------------------------------------------------------------------------
extern "C" void kernel_launch(void* const* d_in, const int* in_sizes, int n_in,
                              void* d_out, int out_size, void* d_ws, size_t ws_size,
                              hipStream_t stream) {
    (void)in_sizes; (void)n_in; (void)out_size; (void)ws_size;
    const float* x  = (const float*)d_in[0];
    const float* Wq = (const float*)d_in[1];
    const float* bq = (const float*)d_in[2];
    const float* Wk = (const float*)d_in[3];
    const float* bk = (const float*)d_in[4];
    const float* Wv = (const float*)d_in[5];
    const float* bv = (const float*)d_in[6];
    const float* Wo = (const float*)d_in[7];
    const float* bo = (const float*)d_in[8];

    const int M = MROWS, N = D_MODEL, Kd = D_MODEL;

    char*  ws  = (char*)d_ws;
    size_t off = 0;
    auto carve = [&](size_t bytes) {
        void* p = ws + off;
        off += (bytes + 255) & ~(size_t)255;
        return p;
    };
    unsigned short* Xb   = (unsigned short*)carve((size_t)M * Kd * 2);
    unsigned short* Wqb  = (unsigned short*)carve((size_t)N * Kd * 2);
    unsigned short* Wkb  = (unsigned short*)carve((size_t)N * Kd * 2);
    unsigned short* Wvb  = (unsigned short*)carve((size_t)N * Kd * 2);
    unsigned short* Wob  = (unsigned short*)carve((size_t)N * Kd * 2);
    unsigned short* Qbuf = (unsigned short*)carve((size_t)Bq * N_HEADS * SEQ * HEADDIM * 2);
    unsigned short* Kbuf = (unsigned short*)carve((size_t)Bq * N_HEADS * SEQ * HEADDIM * 2);
    unsigned short* Vbuf = (unsigned short*)carve((size_t)Bq * N_HEADS * SEQ * HEADDIM * 2);
    unsigned short* Ctxb = (unsigned short*)carve((size_t)M * D_MODEL * 2);

    int nx = M * Kd, nw = N * Kd;
    cvt_f32_bf16<<<(nx + 255) / 256, 256, 0, stream>>>(x,  Xb,  nx);
    cvt_f32_bf16<<<(nw + 255) / 256, 256, 0, stream>>>(Wq, Wqb, nw);
    cvt_f32_bf16<<<(nw + 255) / 256, 256, 0, stream>>>(Wk, Wkb, nw);
    cvt_f32_bf16<<<(nw + 255) / 256, 256, 0, stream>>>(Wv, Wvb, nw);
    cvt_f32_bf16<<<(nw + 255) / 256, 256, 0, stream>>>(Wo, Wob, nw);

    dim3 gg(M / 128, N / 64);
    gemm_wmma<<<gg, 128, 0, stream>>>(Xb, Wqb, bq, Qbuf, M, N, Kd, SEQ, N_HEADS, 0);
    gemm_wmma<<<gg, 128, 0, stream>>>(Xb, Wkb, bk, Kbuf, M, N, Kd, SEQ, N_HEADS, 0);
    gemm_wmma<<<gg, 128, 0, stream>>>(Xb, Wvb, bv, Vbuf, M, N, Kd, SEQ, N_HEADS, 0);

    attn_wmma<<<dim3(SEQ / 64, Bq * N_HEADS), 128, 0, stream>>>(
        Qbuf, Kbuf, Vbuf, Ctxb, SEQ, N_HEADS);

    gemm_wmma<<<gg, 128, 0, stream>>>(Ctxb, Wob, bo, (float*)d_out,
                                      M, N, Kd, SEQ, N_HEADS, 1);
}
